// SNNModel_75986561401182
// MI455X (gfx1250) — matrix operand order repaired and checked by reading, not compiled, not run on hardware
//
#include <hip/hip_runtime.h>

// ---------------------------------------------------------------------------
// SNN (3-layer LIF, 100 steps) for MI455X / gfx1250.
//   * spk1 trajectory depends only on constant cur1 -> precompute all steps;
//     the 100 serialized GEMMs collapse into one [25600,1024]@[1024,1024].
//   * Dominant GEMM uses V_WMMA_F32_16X16X32_BF16 with an error-free weight
//     split (W2 = bf16_hi + bf16_lo, spikes are exactly {0,1} in bf16):
//     ~f32 accuracy at bf16 matrix-core density (8192 MACs/instr vs 1024).
//   * Small GEMMs (layer 1 f32 activations, layer 3 N=10) stay on exact
//     V_WMMA_F32_16X16X4_F32.
// ---------------------------------------------------------------------------

typedef __attribute__((ext_vector_type(2)))  float  v2f;
typedef __attribute__((ext_vector_type(8)))  float  v8f;
typedef __attribute__((ext_vector_type(16))) __bf16 v16bf;

union FragBF {  // 16 bf16 = 32B, loaded as two b128 chunks
  v16bf v;
  uint4 q[2];
};

#define BETA 0.95f
#define THRESH 1.0f
#define NSTEPS 100
#define BATCH 256
#define INDIM 784
#define HID 1024
#define ODIM 10
#define ODIM_PAD 16

// ---------------------------------------------------------------------------
// f32 GEMM: C[M,N] = A[M,K]@B[K,N] + bias. Wave = 32x32 tile (2x2 WMMA frags),
// block = 8 waves (2Mx4N) -> 64x128 workgroup tile. Exact f32 path.
// ---------------------------------------------------------------------------
__global__ void gemm_f32_wmma(const float* __restrict__ A,
                              const float* __restrict__ B,
                              const float* __restrict__ bias,
                              float* __restrict__ C,
                              int M, int N, int K) {
  const int lane = threadIdx.x & 31;
  const int wave = threadIdx.x >> 5;
  const int wm   = wave >> 2;
  const int wn   = wave & 3;
  const int half = lane >> 4;
  const int l16  = lane & 15;

  const int tileM = blockIdx.y * 64  + wm * 32;
  const int tileN = blockIdx.x * 128 + wn * 32;

  v8f acc[2][2] = {};

  for (int k = 0; k < K; k += 4) {
    v2f a[2], b[2];
#pragma unroll
    for (int mi = 0; mi < 2; ++mi) {
      const float* ap = A + (size_t)(tileM + mi * 16 + l16) * K + k + half * 2;
      a[mi] = *(const v2f*)ap;
    }
#pragma unroll
    for (int ni = 0; ni < 2; ++ni) {
      const float* bp = B + (size_t)(k + half * 2) * N + tileN + ni * 16 + l16;
      b[ni][0] = bp[0];
      b[ni][1] = bp[N];
    }
#pragma unroll
    for (int mi = 0; mi < 2; ++mi)
#pragma unroll
      for (int ni = 0; ni < 2; ++ni)
        acc[mi][ni] = __builtin_amdgcn_wmma_f32_16x16x4_f32(
            false, a[mi], false, b[ni], (short)0, acc[mi][ni], false, false);
  }

#pragma unroll
  for (int mi = 0; mi < 2; ++mi)
#pragma unroll
    for (int ni = 0; ni < 2; ++ni) {
      const int n  = tileN + ni * 16 + l16;
      const float bv = bias[n];
#pragma unroll
      for (int r = 0; r < 8; ++r) {
        const int m = tileM + mi * 16 + half * 8 + r;
        C[(size_t)m * N + n] = acc[mi][ni][r] + bv;
      }
    }
}

// ---------------------------------------------------------------------------
// bf16 split GEMM (dominant layer-2 batch):
//   C[M,N] = A_bf16[M,K] @ (Bt_hi + Bt_lo)^T [K,N] + bias
// A is spikes (exact in bf16). Bt_hi/Bt_lo are W2 split, stored TRANSPOSED
// [N,K] so both A and B fragments are contiguous b128 loads:
//   A 16x32: lane half=0 -> K k..k+15; half=1 -> K k+16..k+31 (matches B)
//   B 32x16: VGPR j lanes0-15 = {K=2j,2j+1}, lanes16-31 = {K=16+2j,...}
//     -> per lane: 16 consecutive K at Bt[n*K + k + half*16]
// Wave = 32x32 tile: 2 A frags x 2 n-tiles x {hi,lo} = 8 wmma per K=32.
// ---------------------------------------------------------------------------
__global__ void gemm_bf16split_wmma(const __bf16* __restrict__ A,
                                    const __bf16* __restrict__ Bth,
                                    const __bf16* __restrict__ Btl,
                                    const float* __restrict__ bias,
                                    float* __restrict__ C,
                                    int M, int N, int K) {
  const int lane = threadIdx.x & 31;
  const int wave = threadIdx.x >> 5;
  const int wm   = wave >> 2;
  const int wn   = wave & 3;
  const int half = lane >> 4;
  const int l16  = lane & 15;

  const int tileM = blockIdx.y * 64  + wm * 32;
  const int tileN = blockIdx.x * 128 + wn * 32;

  v8f acc[2][2] = {};

  const __bf16* arow0 = A + (size_t)(tileM + l16) * K + half * 16;
  const __bf16* arow1 = arow0 + (size_t)16 * K;
  const __bf16* bh0 = Bth + (size_t)(tileN + l16) * K + half * 16;
  const __bf16* bh1 = bh0 + (size_t)16 * K;
  const __bf16* bl0 = Btl + (size_t)(tileN + l16) * K + half * 16;
  const __bf16* bl1 = bl0 + (size_t)16 * K;

  for (int k = 0; k < K; k += 32) {
    FragBF a[2], bh[2], bl[2];
    a[0].q[0] = *(const uint4*)(arow0 + k);
    a[0].q[1] = *(const uint4*)(arow0 + k + 8);
    a[1].q[0] = *(const uint4*)(arow1 + k);
    a[1].q[1] = *(const uint4*)(arow1 + k + 8);
    bh[0].q[0] = *(const uint4*)(bh0 + k);
    bh[0].q[1] = *(const uint4*)(bh0 + k + 8);
    bh[1].q[0] = *(const uint4*)(bh1 + k);
    bh[1].q[1] = *(const uint4*)(bh1 + k + 8);
    bl[0].q[0] = *(const uint4*)(bl0 + k);
    bl[0].q[1] = *(const uint4*)(bl0 + k + 8);
    bl[1].q[0] = *(const uint4*)(bl1 + k);
    bl[1].q[1] = *(const uint4*)(bl1 + k + 8);
#pragma unroll
    for (int mi = 0; mi < 2; ++mi)
#pragma unroll
      for (int ni = 0; ni < 2; ++ni) {
        acc[mi][ni] = __builtin_amdgcn_wmma_f32_16x16x32_bf16(
            false, a[mi].v, false, bl[ni].v, (short)0, acc[mi][ni], false, false);
        acc[mi][ni] = __builtin_amdgcn_wmma_f32_16x16x32_bf16(
            false, a[mi].v, false, bh[ni].v, (short)0, acc[mi][ni], false, false);
      }
  }

#pragma unroll
  for (int mi = 0; mi < 2; ++mi)
#pragma unroll
    for (int ni = 0; ni < 2; ++ni) {
      const int n  = tileN + ni * 16 + l16;
      const float bv = bias[n];
#pragma unroll
      for (int r = 0; r < 8; ++r) {
        const int m = tileM + mi * 16 + half * 8 + r;
        C[(size_t)m * N + n] = acc[mi][ni][r] + bv;
      }
    }
}

// ---------------------------------------------------------------------------
// Narrow output GEMM: C[M,16] = A[M,K]@B[K,10] + bias, cols >= 10 zero via
// per-lane select (EXEC stays all-1s for WMMA). Wave = 32x16 tile.
// ---------------------------------------------------------------------------
__global__ void gemm_f32_wmma_n16(const float* __restrict__ A,
                                  const float* __restrict__ B,
                                  const float* __restrict__ bias,
                                  float* __restrict__ C,
                                  int M, int K, int Nvalid, int ldc) {
  const int lane = threadIdx.x & 31;
  const int wave = threadIdx.x >> 5;
  const int half = lane >> 4;
  const int l16  = lane & 15;

  const int tileM = blockIdx.x * 256 + wave * 32;
  const bool ncol = (l16 < Nvalid);

  v8f acc[2] = {};

  for (int k = 0; k < K; k += 4) {
    v2f a[2], b;
#pragma unroll
    for (int mi = 0; mi < 2; ++mi) {
      const float* ap = A + (size_t)(tileM + mi * 16 + l16) * K + k + half * 2;
      a[mi] = *(const v2f*)ap;
    }
    const float* bp = B + (size_t)(k + half * 2) * Nvalid + l16;
    b[0] = ncol ? bp[0] : 0.0f;
    b[1] = ncol ? bp[Nvalid] : 0.0f;
#pragma unroll
    for (int mi = 0; mi < 2; ++mi)
      acc[mi] = __builtin_amdgcn_wmma_f32_16x16x4_f32(
          false, a[mi], false, b, (short)0, acc[mi], false, false);
  }

  const float bv = ncol ? bias[l16] : 0.0f;
#pragma unroll
  for (int mi = 0; mi < 2; ++mi)
#pragma unroll
    for (int r = 0; r < 8; ++r) {
      const int m = tileM + mi * 16 + half * 8 + r;
      C[(size_t)m * ldc + l16] = acc[mi][r] + bv;
    }
}

// ---------------------------------------------------------------------------
// W[K,N] f32 -> transposed bf16 error-split pair Th/Tl[N,K].
// ---------------------------------------------------------------------------
__global__ void split_transpose_bf16(const float* __restrict__ W,
                                     __bf16* __restrict__ Th,
                                     __bf16* __restrict__ Tl,
                                     int K, int N) {
  const int idx = blockIdx.x * blockDim.x + threadIdx.x;
  if (idx >= K * N) return;
  const int kk = idx / N;
  const int n  = idx % N;
  const float w  = W[idx];
  const __bf16 hi = (__bf16)w;
  const __bf16 lo = (__bf16)(w - (float)hi);
  Th[(size_t)n * K + kk] = hi;
  Tl[(size_t)n * K + kk] = lo;
}

// ---------------------------------------------------------------------------
// LIF scans (snntorch order: reset from previous mem, then decay+integrate,
// then fire from updated mem).
// ---------------------------------------------------------------------------
__global__ void lif_seq_const_bf16(const float* __restrict__ cur,
                                   __bf16* __restrict__ spk_out, int BH) {
  const int idx = blockIdx.x * blockDim.x + threadIdx.x;
  if (idx >= BH) return;
  const float c = cur[idx];
  float mem = 0.0f;
  for (int t = 0; t < NSTEPS; ++t) {
    const float reset = (mem - THRESH > 0.0f) ? 1.0f : 0.0f;
    mem = BETA * mem + c - reset * THRESH;
    spk_out[(size_t)t * BH + idx] = (__bf16)((mem - THRESH > 0.0f) ? 1.0f : 0.0f);
  }
}

__global__ void lif_seq_inplace(float* __restrict__ buf, int BH) {
  const int idx = blockIdx.x * blockDim.x + threadIdx.x;
  if (idx >= BH) return;
  float mem = 0.0f;
  for (int t = 0; t < NSTEPS; ++t) {
    const float c = buf[(size_t)t * BH + idx];
    const float reset = (mem - THRESH > 0.0f) ? 1.0f : 0.0f;
    mem = BETA * mem + c - reset * THRESH;
    buf[(size_t)t * BH + idx] = (mem - THRESH > 0.0f) ? 1.0f : 0.0f;
  }
}

__global__ void lif_seq_out(const float* __restrict__ cur3,  // [T*B, ODIM_PAD]
                            float* __restrict__ out) {       // [T, B, ODIM]
  const int idx = blockIdx.x * blockDim.x + threadIdx.x;
  if (idx >= BATCH * ODIM) return;
  const int b = idx / ODIM;
  const int o = idx % ODIM;
  float mem = 0.0f;
  for (int t = 0; t < NSTEPS; ++t) {
    const float c = cur3[((size_t)t * BATCH + b) * ODIM_PAD + o];
    const float reset = (mem - THRESH > 0.0f) ? 1.0f : 0.0f;
    mem = BETA * mem + c - reset * THRESH;
    out[(size_t)t * BATCH * ODIM + idx] = (mem - THRESH > 0.0f) ? 1.0f : 0.0f;
  }
}

// ---------------------------------------------------------------------------
extern "C" void kernel_launch(void* const* d_in, const int* in_sizes, int n_in,
                              void* d_out, int out_size, void* d_ws, size_t ws_size,
                              hipStream_t stream) {
  const float* x  = (const float*)d_in[0];   // [256,784]
  const float* W1 = (const float*)d_in[1];   // [784,1024]
  const float* b1 = (const float*)d_in[2];   // [1024]
  const float* W2 = (const float*)d_in[3];   // [1024,1024]
  const float* b2 = (const float*)d_in[4];   // [1024]
  const float* W3 = (const float*)d_in[5];   // [1024,10]
  const float* b3 = (const float*)d_in[6];   // [10]
  float* out = (float*)d_out;                // [100,256,10]

  const size_t BH  = (size_t)BATCH * HID;    // 262144
  const size_t TBH = (size_t)NSTEPS * BH;    // 26,214,400

  // Workspace layout (all offsets multiples of 256 B), ~164 MB total:
  char* ws = (char*)d_ws;
  float*  cur1 = (float*)(ws);                       //   1.00 MB f32 [256,1024]
  __bf16* spk1 = (__bf16*)(ws + 1048576);            //  52.43 MB bf16 [25600,1024]
  float*  cur2 = (float*)(ws + 53477376);            // 104.86 MB f32 [25600,1024]
  float*  cur3 = (float*)(ws + 158334976);           //   1.64 MB f32 [25600,16]
  __bf16* w2h  = (__bf16*)(ws + 159973376);          //   2.00 MB bf16 [1024,1024] (W2^T hi)
  __bf16* w2l  = (__bf16*)(ws + 162070528);          //   2.00 MB bf16 [1024,1024] (W2^T lo)
  (void)ws_size; (void)in_sizes; (void)n_in; (void)out_size;

  // P0: W2 -> transposed bf16 error-split pair (reused 25600x in K3)
  split_transpose_bf16<<<(HID * HID + 255) / 256, 256, 0, stream>>>(
      W2, w2h, w2l, HID, HID);

  // K1: cur1[256,1024] = x @ W1 + b1  (exact f32 WMMA, K=784)
  gemm_f32_wmma<<<dim3(HID / 128, BATCH / 64), 256, 0, stream>>>(
      x, W1, b1, cur1, BATCH, HID, INDIM);

  // K2: full spk1 trajectory, stored bf16 (spikes are exactly {0,1})
  lif_seq_const_bf16<<<(int)((BH + 255) / 256), 256, 0, stream>>>(
      cur1, spk1, (int)BH);

  // K3: cur2_all[25600,1024] = spk1_all @ (W2hi + W2lo) + b2  -- dominant GEMM
  gemm_bf16split_wmma<<<dim3(HID / 128, (NSTEPS * BATCH) / 64), 256, 0, stream>>>(
      spk1, w2h, w2l, b2, cur2, NSTEPS * BATCH, HID, HID);

  // K4: mem2 scan, in place (cur2 -> spk2, f32)
  lif_seq_inplace<<<(int)((BH + 255) / 256), 256, 0, stream>>>(cur2, (int)BH);

  // K5: cur3_all[25600,16(pad)] = spk2_all @ W3 + b3  (exact f32 WMMA)
  gemm_f32_wmma_n16<<<(NSTEPS * BATCH) / 256, 256, 0, stream>>>(
      cur2, W3, b3, cur3, NSTEPS * BATCH, HID, ODIM, ODIM_PAD);

  // K6: mem3 scan -> spk3 output [100,256,10]
  lif_seq_out<<<(BATCH * ODIM + 255) / 256, 256, 0, stream>>>(cur3, out);
}
